// Longformer_60988535603481
// MI455X (gfx1250) — matrix operand order repaired and checked by reading, not compiled, not run on hardware
//
#include <hip/hip_runtime.h>
#include <hip/hip_bf16.h>
#include <math.h>

// ---------------------------------------------------------------------------
// Longformer forward for MI455X (gfx1250, wave32, WMMA + async global->LDS).
// ---------------------------------------------------------------------------

#define B_  8
#define S_  768
#define H_  768
#define NH_ 12
#define L_  12
#define D_  64
#define W_  128
#define FF_ 3072
#define MROWS (B_ * S_)        // 6144
#define LN_EPS 1e-12f

typedef __bf16 bf16_t;
typedef __attribute__((ext_vector_type(16))) __bf16 v16bf;
typedef __attribute__((ext_vector_type(8)))  float  v8f;
typedef __attribute__((ext_vector_type(4)))  int    v4i_;

// ---------------------------------------------------------------------------
// Async global->LDS (16B per lane), ASYNCcnt tracked.
// Builtin signature (from hipcc diagnostics): param0 = v4i AS(1)*, so we pass
// typed, address-space-qualified vector pointers.
// ---------------------------------------------------------------------------
#if defined(__has_builtin)
# if __has_builtin(__builtin_amdgcn_global_load_async_to_lds_b128)
#  define HAVE_ASYNC_BUILTIN 1
# endif
# if __has_builtin(__builtin_amdgcn_s_wait_asynccnt)
#  define HAVE_WAIT_BUILTIN 1
# endif
#endif

__device__ __forceinline__ void async16(const bf16_t* g, bf16_t* l) {
#if defined(HAVE_ASYNC_BUILTIN)
    __builtin_amdgcn_global_load_async_to_lds_b128(
        (__attribute__((address_space(1))) v4i_*)(unsigned long long)g,
        (__attribute__((address_space(3))) v4i_*)(unsigned int)(unsigned long long)l,
        0, 0);
#else
    unsigned int lds = (unsigned int)(unsigned long long)l;
    unsigned long long ga = (unsigned long long)g;
    asm volatile("global_load_async_to_lds_b128 %0, %1, off"
                 :: "v"(lds), "v"(ga) : "memory");
#endif
}

__device__ __forceinline__ void wait_async() {
#if defined(HAVE_WAIT_BUILTIN)
    __builtin_amdgcn_s_wait_asynccnt(0);
#else
    asm volatile("s_wait_asynccnt 0" ::: "memory");
#endif
}

// ---------------------------------------------------------------------------
// WMMA fragment loaders (wave32), layouts per CDNA5 ISA.
// A: 16x32 bf16 tile, row-major [m][k]. lane: m=L&15, kb=(L>>4)*8;
//    elems 0..7 -> K=kb..kb+7 ; elems 8..15 -> K=kb+16..kb+23.
// B: 32x16 tile stored transposed [n][k]. lane: n=L&15, kb=(L>>4)*8;
//    elems 0..7 -> K=kb..kb+7 ; elems 8..15 -> K=kb+16..kb+23.
// ---------------------------------------------------------------------------
__device__ __forceinline__ v16bf load_frag_a(const bf16_t* base, int ld) {
    int lane = threadIdx.x & 31;
    const bf16_t* p = base + (lane & 15) * ld + ((lane >> 4) << 3);
    union { v16bf v; uint4 u[2]; } r;
    r.u[0] = *(const uint4*)(p);
    r.u[1] = *(const uint4*)(p + 16);
    return r.v;
}

__device__ __forceinline__ v16bf load_frag_b(const bf16_t* base, int ld) {
    int lane = threadIdx.x & 31;
    const bf16_t* p = base + (lane & 15) * ld + ((lane >> 4) << 3);
    union { v16bf v; uint4 u[2]; } r;
    r.u[0] = *(const uint4*)(p);
    r.u[1] = *(const uint4*)(p + 16);
    return r.v;
}

__device__ __forceinline__ v8f wmma_bf16(v16bf a, v16bf b, v8f c) {
    return __builtin_amdgcn_wmma_f32_16x16x32_bf16(
        false, a, false, b, (short)0, c, false, false);
}

// ---------------------------------------------------------------------------
// Weight prep: src [K][N] fp32 -> dst [N][K] bf16 (per layer, blockIdx.z)
// ---------------------------------------------------------------------------
__global__ __launch_bounds__(256) void transpose_cvt(
    const float* __restrict__ src, bf16_t* __restrict__ dst, int K, int N)
{
    __shared__ float t[32][33];
    const int l = blockIdx.z;
    const float* s = src + (size_t)l * K * N;
    bf16_t* d = dst + (size_t)l * K * N;
    const int k0 = blockIdx.y * 32, n0 = blockIdx.x * 32;
    const int tx = threadIdx.x & 31, ty = threadIdx.x >> 5;   // 8 rows/pass
    for (int i = 0; i < 32; i += 8)
        t[ty + i][tx] = s[(size_t)(k0 + ty + i) * N + n0 + tx];
    __syncthreads();
    for (int i = 0; i < 32; i += 8)
        d[(size_t)(n0 + ty + i) * K + k0 + tx] = (bf16_t)t[tx][ty + i];
}

// ---------------------------------------------------------------------------
// GEMM: C[M,N] = A[M,K](fp32) * Bt[N,K](bf16, pre-transposed) + bias[N]
// Output fp32 (C) or bf16 (Cbf). Block tile 128x128, K step 32,
// double-buffered LDS; B tiles staged via async global->LDS DMA.
// ---------------------------------------------------------------------------
#define GA_LD 40   // 32 + 8 pad; 80B rows keep 16B fragment alignment

__global__ __launch_bounds__(256) void gemm_bf16_wmma(
    const float* __restrict__ A, const bf16_t* __restrict__ Bt,
    const float* __restrict__ bias, float* __restrict__ C,
    bf16_t* __restrict__ Cbf, int M, int N, int K)
{
    __shared__ __align__(16) bf16_t As[2][128 * GA_LD];
    __shared__ __align__(16) bf16_t Bs[2][128 * GA_LD];

    const int bm = blockIdx.y * 128;
    const int bn = blockIdx.x * 128;
    const int tid  = threadIdx.x;
    const int wave = tid >> 5;
    const int wm = (wave >> 1) * 32;   // 0,32,64,96
    const int wn = (wave & 1) * 64;    // 0,64
    const int lane = tid & 31;
    const int n_in = lane & 15;
    const int hl   = lane >> 4;

    v8f acc[2][4];
    for (int r = 0; r < 2; ++r)
        for (int c = 0; c < 4; ++c)
            for (int e = 0; e < 8; ++e) acc[r][c][e] = 0.0f;

    auto stage_a = [&](int st, int k0) {     // 128x32 fp32 -> bf16
        for (int i = tid; i < 1024; i += 256) {
            int idx = i << 2;
            int row = idx >> 5, col = idx & 31;
            float4 a4 = *(const float4*)(A + (size_t)(bm + row) * K + k0 + col);
            bf16_t* d = &As[st][row * GA_LD + col];
            d[0] = (bf16_t)a4.x; d[1] = (bf16_t)a4.y;
            d[2] = (bf16_t)a4.z; d[3] = (bf16_t)a4.w;
        }
    };
    auto stage_b = [&](int st, int k0) {     // 128 rows x 64B, async DMA
        for (int i = tid; i < 512; i += 256) {
            int row = i >> 2, cp = (i & 3) << 3;
            async16(Bt + (size_t)(bn + row) * K + k0 + cp,
                    &Bs[st][row * GA_LD + cp]);
        }
    };

    stage_b(0, 0);
    stage_a(0, 0);
    wait_async();
    __syncthreads();

    int st = 0;
    for (int k0 = 0; k0 < K; k0 += 32) {
        const int kn = k0 + 32;
        if (kn < K) { stage_b(st ^ 1, kn); stage_a(st ^ 1, kn); }

        v16bf af[2], bfr[4];
        for (int r = 0; r < 2; ++r) af[r]  = load_frag_a(&As[st][(wm + r * 16) * GA_LD], GA_LD);
        for (int c = 0; c < 4; ++c) bfr[c] = load_frag_b(&Bs[st][(wn + c * 16) * GA_LD], GA_LD);
        for (int r = 0; r < 2; ++r)
            for (int c = 0; c < 4; ++c)
                acc[r][c] = wmma_bf16(af[r], bfr[c], acc[r][c]);

        wait_async();
        __syncthreads();
        st ^= 1;
    }

    float bsv[4];
    for (int c = 0; c < 4; ++c) bsv[c] = bias[bn + wn + c * 16 + n_in];

    if (Cbf) {
        for (int r = 0; r < 2; ++r)
            for (int c = 0; c < 4; ++c)
                for (int e = 0; e < 8; ++e) {
                    int m = bm + wm + r * 16 + e + hl * 8;
                    int n = bn + wn + c * 16 + n_in;
                    Cbf[(size_t)m * N + n] = (bf16_t)(acc[r][c][e] + bsv[c]);
                }
    } else {
        for (int r = 0; r < 2; ++r)
            for (int c = 0; c < 4; ++c)
                for (int e = 0; e < 8; ++e) {
                    int m = bm + wm + r * 16 + e + hl * 8;
                    int n = bn + wn + c * 16 + n_in;
                    C[(size_t)m * N + n] = acc[r][c][e] + bsv[c];
                }
    }
}

// ---------------------------------------------------------------------------
// Sliding-window attention (bf16 Q/K/V in, fp32 O out), flash-style softmax.
// Q and K tiles staged via async DMA (straight bf16 copies); 1/8 scale is
// applied to the scores after the QK^T WMMA instead of to Q.
// ---------------------------------------------------------------------------
#define AQ_LD 72    // 64 + 8
#define PV_LD 136   // 128 + 8

__global__ __launch_bounds__(256) void attn_window(
    const bf16_t* __restrict__ Qg, const bf16_t* __restrict__ Kg,
    const bf16_t* __restrict__ Vg, float* __restrict__ O)
{
    const int c    = blockIdx.x;   // chunk 0..5
    const int head = blockIdx.y;
    const int bz   = blockIdx.z;

    __shared__ __align__(16) bf16_t Aq[128 * AQ_LD];
    // Overlay: phase 1 = Kt [128][AQ_LD]; phase 2 = Vt [64][PV_LD] + P [8][16][PV_LD]
    __shared__ __align__(16) unsigned char smem[(64 * PV_LD + 8 * 16 * PV_LD) * 2];
    bf16_t* Kt = (bf16_t*)smem;
    bf16_t* Vt = (bf16_t*)smem;
    bf16_t* Pl = (bf16_t*)(smem + 64 * PV_LD * 2);

    const int tid  = threadIdx.x;
    const int wave = tid >> 5;
    const int lane = tid & 31;
    const int n_in = lane & 15;
    const int hl   = lane >> 4;

    // Q chunk -> LDS, pure async copy (rows of 128B, 8 x 16B chunks)
    const bf16_t* Qbase = Qg + ((size_t)(bz * S_ + c * 128) * H_ + head * D_);
    for (int i = tid; i < 1024; i += 256) {
        int row = i >> 3, cp = (i & 7) << 3;
        async16(Qbase + (size_t)row * H_ + cp, &Aq[row * AQ_LD + cp]);
    }
    wait_async();
    __syncthreads();

    v16bf afQ[2];
    afQ[0] = load_frag_a(&Aq[(wave * 16) * AQ_LD + 0],  AQ_LD);
    afQ[1] = load_frag_a(&Aq[(wave * 16) * AQ_LD + 32], AQ_LD);

    float m_row[8], l_row[8];
    v8f acc_o[4];
    for (int r = 0; r < 8; ++r) { m_row[r] = -1e30f; l_row[r] = 0.0f; }
    for (int t = 0; t < 4; ++t)
        for (int e = 0; e < 8; ++e) acc_o[t][e] = 0.0f;

    for (int kt = 0; kt < 3; ++kt) {
        const int key0 = (c - 1 + kt) * 128;

        // ---- stage K tile [key][d]: async for in-range rows, zero otherwise
        for (int i = tid; i < 1024; i += 256) {
            int row = i >> 3, cp = (i & 7) << 3;
            int kr = key0 + row;
            bf16_t* dst = &Kt[row * AQ_LD + cp];
            if (kr >= 0 && kr < S_)
                async16(Kg + ((size_t)(bz * S_ + kr) * H_ + head * D_ + cp), dst);
            else {
                uint4 z = make_uint4(0u, 0u, 0u, 0u);
                *(uint4*)dst = z;
            }
        }
        wait_async();
        __syncthreads();

        // ---- scores: 8 key-subtiles x 2 WMMA k-steps (D=64)
        v8f sc[8];
        for (int j = 0; j < 8; ++j) {
            for (int e = 0; e < 8; ++e) sc[j][e] = 0.0f;
            v16bf bk0 = load_frag_b(&Kt[(j * 16) * AQ_LD + 0],  AQ_LD);
            v16bf bk1 = load_frag_b(&Kt[(j * 16) * AQ_LD + 32], AQ_LD);
            sc[j] = wmma_bf16(afQ[0], bk0, sc[j]);
            sc[j] = wmma_bf16(afQ[1], bk1, sc[j]);
        }
        __syncthreads();   // all waves done reading Kt (aliased by Vt/Pl)

        // ---- scale + mask + online softmax
        float newm[8];
        for (int r = 0; r < 8; ++r) newm[r] = m_row[r];
        for (int j = 0; j < 8; ++j)
            for (int e = 0; e < 8; ++e) {
                int qa = c * 128 + wave * 16 + e + hl * 8;
                int ka = key0 + j * 16 + n_in;
                int rel = ka - qa;
                bool valid = (ka >= 0) && (ka < S_) && (rel <= W_) && (rel >= -W_);
                float s = valid ? sc[j][e] * 0.125f : -1e30f;
                sc[j][e] = s;
                newm[e] = fmaxf(newm[e], s);
            }
        for (int r = 0; r < 8; ++r)
            for (int off = 1; off < 16; off <<= 1)
                newm[r] = fmaxf(newm[r], __shfl_xor(newm[r], off, 32));

        float scale[8], rs[8];
        for (int r = 0; r < 8; ++r) {
            scale[r] = __expf(m_row[r] - newm[r]);
            m_row[r] = newm[r];
            rs[r] = 0.0f;
        }
        for (int j = 0; j < 8; ++j)
            for (int e = 0; e < 8; ++e) {
                float p = __expf(sc[j][e] - m_row[e]);
                sc[j][e] = p;
                rs[e] += p;
            }
        for (int r = 0; r < 8; ++r)
            for (int off = 1; off < 16; off <<= 1)
                rs[r] += __shfl_xor(rs[r], off, 32);
        for (int r = 0; r < 8; ++r) l_row[r] = l_row[r] * scale[r] + rs[r];
        for (int t = 0; t < 4; ++t)
            for (int e = 0; e < 8; ++e) acc_o[t][e] *= scale[e];

        // ---- phase 2: P -> LDS (A layout) and V tile transposed [d][key]
        bf16_t* Pw = Pl + wave * 16 * PV_LD;
        for (int j = 0; j < 8; ++j)
            for (int e = 0; e < 8; ++e)
                Pw[(e + hl * 8) * PV_LD + j * 16 + n_in] = (bf16_t)sc[j][e];
        for (int i = tid; i < 1024; i += 256) {
            int row = i >> 3, dp = (i & 7) << 3;
            int kr = key0 + row;
            union { uint4 u; bf16_t h[8]; } vv;
            vv.u = make_uint4(0u, 0u, 0u, 0u);
            if (kr >= 0 && kr < S_)
                vv.u = *(const uint4*)(Vg + ((size_t)(bz * S_ + kr) * H_ + head * D_ + dp));
            for (int j = 0; j < 8; ++j)
                Vt[(dp + j) * PV_LD + row] = vv.h[j];
        }
        __syncthreads();

        // ---- PV: O[16x64] += P[16x128] * V[128x64]
        v16bf afP[4];
        for (int kk = 0; kk < 4; ++kk)
            afP[kk] = load_frag_a(Pw + kk * 32, PV_LD);
        for (int t = 0; t < 4; ++t)
            for (int kk = 0; kk < 4; ++kk) {
                v16bf bv = load_frag_b(&Vt[(t * 16) * PV_LD + kk * 32], PV_LD);
                acc_o[t] = wmma_bf16(afP[kk], bv, acc_o[t]);
            }
        __syncthreads();   // done with Vt/Pl before next K stage
    }

    float inv[8];
    for (int r = 0; r < 8; ++r) inv[r] = 1.0f / l_row[r];
    for (int t = 0; t < 4; ++t)
        for (int e = 0; e < 8; ++e) {
            int qa = c * 128 + wave * 16 + e + hl * 8;
            O[((size_t)(bz * S_ + qa)) * H_ + head * D_ + t * 16 + n_in] = acc_o[t][e] * inv[e];
        }
}

// ---------------------------------------------------------------------------
// Row kernels
// ---------------------------------------------------------------------------
__global__ __launch_bounds__(256) void embed_ln_kernel(
    const float* __restrict__ emb, const float* __restrict__ pos,
    const float* __restrict__ tte, const float* __restrict__ g,
    const float* __restrict__ b, float* __restrict__ out)
{
    const int row = blockIdx.x;
    const int s   = row % S_;
    const int t   = threadIdx.x;
    __shared__ float red[256];

    float v[3];
    float sum = 0.f;
    for (int i = 0; i < 3; ++i) {
        int col = t + i * 256;
        v[i] = emb[(size_t)row * H_ + col] + pos[(size_t)(s + 2) * H_ + col] + tte[col];
        sum += v[i];
    }
    red[t] = sum; __syncthreads();
    for (int off = 128; off > 0; off >>= 1) { if (t < off) red[t] += red[t + off]; __syncthreads(); }
    float mu = red[0] * (1.0f / H_); __syncthreads();
    float q = 0.f;
    for (int i = 0; i < 3; ++i) { float d = v[i] - mu; q += d * d; }
    red[t] = q; __syncthreads();
    for (int off = 128; off > 0; off >>= 1) { if (t < off) red[t] += red[t + off]; __syncthreads(); }
    float rsv = rsqrtf(red[0] * (1.0f / H_) + LN_EPS);
    for (int i = 0; i < 3; ++i) {
        int col = t + i * 256;
        out[(size_t)row * H_ + col] = (v[i] - mu) * rsv * g[col] + b[col];
    }
}

__global__ __launch_bounds__(256) void add_ln_kernel(
    const float* __restrict__ resid, const float* __restrict__ delta,
    const float* __restrict__ g, const float* __restrict__ b,
    float* __restrict__ out)
{
    const int row = blockIdx.x;
    const int t   = threadIdx.x;
    __shared__ float red[256];

    float v[3];
    float sum = 0.f;
    for (int i = 0; i < 3; ++i) {
        int col = t + i * 256;
        v[i] = resid[(size_t)row * H_ + col] + delta[(size_t)row * H_ + col];
        sum += v[i];
    }
    red[t] = sum; __syncthreads();
    for (int off = 128; off > 0; off >>= 1) { if (t < off) red[t] += red[t + off]; __syncthreads(); }
    float mu = red[0] * (1.0f / H_); __syncthreads();
    float q = 0.f;
    for (int i = 0; i < 3; ++i) { float d = v[i] - mu; q += d * d; }
    red[t] = q; __syncthreads();
    for (int off = 128; off > 0; off >>= 1) { if (t < off) red[t] += red[t + off]; __syncthreads(); }
    float rsv = rsqrtf(red[0] * (1.0f / H_) + LN_EPS);
    for (int i = 0; i < 3; ++i) {
        int col = t + i * 256;
        out[(size_t)row * H_ + col] = (v[i] - mu) * rsv * g[col] + b[col];
    }
}

__global__ __launch_bounds__(256) void gelu_kernel(float* __restrict__ x, long n)
{
    long i = (long)blockIdx.x * blockDim.x + threadIdx.x;
    long stride = (long)gridDim.x * blockDim.x;
    for (; i < n; i += stride) {
        float v = x[i];
        x[i] = 0.5f * v * (1.0f + erff(v * 0.70710678118654752f));
    }
}

__global__ __launch_bounds__(256) void gate_kernel(
    const float* __restrict__ x, const float* __restrict__ gw,
    const float* __restrict__ gb, float* __restrict__ out)
{
    const int row = blockIdx.x;
    const int t   = threadIdx.x;
    __shared__ float red[256];
    float sum = 0.f;
    for (int i = 0; i < 3; ++i) {
        int col = t + i * 256;
        sum += x[(size_t)row * H_ + col] * gw[col];
    }
    red[t] = sum; __syncthreads();
    for (int off = 128; off > 0; off >>= 1) { if (t < off) red[t] += red[t + off]; __syncthreads(); }
    if (t == 0) out[row] = red[0] + gb[0];
}

__global__ __launch_bounds__(256) void copy_kernel(
    const float* __restrict__ src, float* __restrict__ dst, long n)
{
    long i = (long)blockIdx.x * blockDim.x + threadIdx.x;
    long stride = (long)gridDim.x * blockDim.x;
    for (; i < n; i += stride) dst[i] = src[i];
}

// ---------------------------------------------------------------------------
// Host launcher
// ---------------------------------------------------------------------------
extern "C" void kernel_launch(void* const* d_in, const int* in_sizes, int n_in,
                              void* d_out, int out_size, void* d_ws, size_t ws_size,
                              hipStream_t stream)
{
    const float* emb      = (const float*)d_in[0];
    const float* pos_emb  = (const float*)d_in[2];
    const float* tte      = (const float*)d_in[3];
    const float* eg       = (const float*)d_in[4];
    const float* eb       = (const float*)d_in[5];
    const float* Wq       = (const float*)d_in[6];
    const float* bq       = (const float*)d_in[7];
    const float* Wk       = (const float*)d_in[8];
    const float* bk       = (const float*)d_in[9];
    const float* Wv       = (const float*)d_in[10];
    const float* bv       = (const float*)d_in[11];
    const float* Wo       = (const float*)d_in[12];
    const float* bo       = (const float*)d_in[13];
    const float* g1       = (const float*)d_in[14];
    const float* b1       = (const float*)d_in[15];
    const float* Wi       = (const float*)d_in[16];
    const float* bi       = (const float*)d_in[17];
    const float* Wo2      = (const float*)d_in[18];
    const float* bo2      = (const float*)d_in[19];
    const float* g2       = (const float*)d_in[20];
    const float* b2       = (const float*)d_in[21];
    const float* gate_w   = (const float*)d_in[22];
    const float* gate_b   = (const float*)d_in[23];

    float* out = (float*)d_out;

    // Workspace layout
    const size_t NX = (size_t)MROWS * H_;         // 4718592
    float* X  = (float*)d_ws;
    float* Ab = X  + NX;
    float* T1 = Ab + NX;
    float* T2 = T1 + NX;                          // MROWS x FF_
    bf16_t* Qb   = (bf16_t*)(T2 + (size_t)MROWS * FF_);
    bf16_t* Kb   = Qb + NX;
    bf16_t* Vb   = Kb + NX;
    bf16_t* WqB  = Vb + NX;                       // [N][K] bf16, per layer
    bf16_t* WkB  = WqB  + (size_t)L_ * H_ * H_;
    bf16_t* WvB  = WkB  + (size_t)L_ * H_ * H_;
    bf16_t* WoB  = WvB  + (size_t)L_ * H_ * H_;
    bf16_t* WiB  = WoB  + (size_t)L_ * H_ * H_;   // [FF][H]
    bf16_t* Wo2B = WiB  + (size_t)L_ * H_ * FF_;  // [H][FF]

    dim3 blk(256);

    // Weight prep: transpose + convert to bf16 once per launch
    transpose_cvt<<<dim3(H_ / 32,  H_ / 32,  L_), blk, 0, stream>>>(Wq,  WqB,  H_,  H_);
    transpose_cvt<<<dim3(H_ / 32,  H_ / 32,  L_), blk, 0, stream>>>(Wk,  WkB,  H_,  H_);
    transpose_cvt<<<dim3(H_ / 32,  H_ / 32,  L_), blk, 0, stream>>>(Wv,  WvB,  H_,  H_);
    transpose_cvt<<<dim3(H_ / 32,  H_ / 32,  L_), blk, 0, stream>>>(Wo,  WoB,  H_,  H_);
    transpose_cvt<<<dim3(FF_ / 32, H_ / 32,  L_), blk, 0, stream>>>(Wi,  WiB,  H_,  FF_);
    transpose_cvt<<<dim3(H_ / 32,  FF_ / 32, L_), blk, 0, stream>>>(Wo2, Wo2B, FF_, H_);

    embed_ln_kernel<<<MROWS, blk, 0, stream>>>(emb, pos_emb, tte, eg, eb, X);

    const dim3 gHH(H_ / 128, MROWS / 128);        // (6,48)
    const dim3 gHF(FF_ / 128, MROWS / 128);       // (24,48)

    for (int l = 0; l < L_; ++l) {
        const bf16_t* WqB_l  = WqB  + (size_t)l * H_ * H_;
        const bf16_t* WkB_l  = WkB  + (size_t)l * H_ * H_;
        const bf16_t* WvB_l  = WvB  + (size_t)l * H_ * H_;
        const bf16_t* WoB_l  = WoB  + (size_t)l * H_ * H_;
        const bf16_t* WiB_l  = WiB  + (size_t)l * H_ * FF_;
        const bf16_t* Wo2B_l = Wo2B + (size_t)l * H_ * FF_;

        gemm_bf16_wmma<<<gHH, blk, 0, stream>>>(X, WqB_l, bq + (size_t)l * H_, nullptr, Qb, MROWS, H_, H_);
        gemm_bf16_wmma<<<gHH, blk, 0, stream>>>(X, WkB_l, bk + (size_t)l * H_, nullptr, Kb, MROWS, H_, H_);
        gemm_bf16_wmma<<<gHH, blk, 0, stream>>>(X, WvB_l, bv + (size_t)l * H_, nullptr, Vb, MROWS, H_, H_);

        attn_window<<<dim3(S_ / 128, NH_, B_), blk, 0, stream>>>(Qb, Kb, Vb, Ab);

        gemm_bf16_wmma<<<gHH, blk, 0, stream>>>(Ab, WoB_l, bo + (size_t)l * H_, T1, nullptr, MROWS, H_, H_);
        add_ln_kernel<<<MROWS, blk, 0, stream>>>(X, T1, g1 + (size_t)l * H_, b1 + (size_t)l * H_, X);

        gemm_bf16_wmma<<<gHF, blk, 0, stream>>>(X, WiB_l, bi + (size_t)l * FF_, T2, nullptr, MROWS, FF_, H_);
        gelu_kernel<<<4096, blk, 0, stream>>>(T2, (long)MROWS * FF_);
        gemm_bf16_wmma<<<gHH, blk, 0, stream>>>(T2, Wo2B_l, bo2 + (size_t)l * H_, T1, nullptr, MROWS, H_, FF_);
        add_ln_kernel<<<MROWS, blk, 0, stream>>>(X, T1, g2 + (size_t)l * H_, b2 + (size_t)l * H_, X);
    }

    copy_kernel<<<2048, blk, 0, stream>>>(X, out, (long)NX);
    gate_kernel<<<MROWS, blk, 0, stream>>>(X, gate_w, gate_b, out + NX);
}
